// RNN_5385888989322
// MI455X (gfx1250) — matrix operand order repaired and checked by reading, not compiled, not run on hardware
//
#include <hip/hip_runtime.h>
#include <hip/hip_bf16.h>

#define SEQ   4096
#define BATCH 128
#define IN_F  64
#define HID   128

typedef __attribute__((ext_vector_type(16))) __bf16 v16bf;
typedef __attribute__((ext_vector_type(8)))  float  v8f;

#if __has_builtin(__builtin_amdgcn_tanhf)
#define TANHF(x) __builtin_amdgcn_tanhf(x)
#elif __has_builtin(__builtin_amdgcn_tanh_f32)
#define TANHF(x) __builtin_amdgcn_tanh_f32(x)
#else
#define TANHF(x) tanhf(x)
#endif

// ---------------------------------------------------------------------------
// Phase 1: xproj = x @ W1^T + b1  written into d_out's h_seq region.
// M = SEQ*BATCH rows, K = 64, N = HID = 128.
// One WG = 8 waves; each wave computes a 16-row x 128-col tile (16 WMMAs).
// ---------------------------------------------------------------------------
__global__ __launch_bounds__(256) void xproj_kernel(
    const float* __restrict__ x, const float* __restrict__ W1,
    const float* __restrict__ b1, float* __restrict__ out)
{
    // W1 as B-matrix fragments: [nt][kt][lane][elem], bf16, 16 KB
    __shared__ __align__(32) __bf16 w1f[8][2][32][16];
    const int tid = threadIdx.x, lane = tid & 31, wave = tid >> 5;

    // Stage W1 -> LDS in B-fragment layout (B(k,n) = W1[n][k])
    for (int idx = tid; idx < 8 * 2 * 32; idx += 256) {
        const int nt = idx >> 6, kt = (idx >> 5) & 1, l = idx & 31;
        const int n  = nt * 16 + (l & 15);
        const int kb = kt * 32 + (l >> 4) * 16;   // B frag: k = e + 16*half
        const float* src = W1 + n * IN_F + kb;
        #pragma unroll
        for (int e = 0; e < 16; ++e) w1f[nt][kt][l][e] = (__bf16)src[e];
    }
    __syncthreads();

    const int  m    = lane & 15, half = lane >> 4;
    const long rbase = ((long)blockIdx.x * 8 + wave) * 16;

    // Build A fragments (x rows, bf16) directly from global: k-pairs contiguous
    v16bf af[2];
    #pragma unroll
    for (int kt = 0; kt < 2; ++kt) {
        const float* row = x + (rbase + m) * IN_F + kt * 32;
        #pragma unroll
        for (int e2 = 0; e2 < 8; ++e2) {
            const int e = 2 * e2;
            const int k = e + 16 * half;
            const float2 v2 = *(const float2*)(row + k);
            af[kt][e]     = (__bf16)v2.x;
            af[kt][e + 1] = (__bf16)v2.y;
        }
    }

    #pragma unroll
    for (int nt = 0; nt < 8; ++nt) {
        v8f acc = {0.f, 0.f, 0.f, 0.f, 0.f, 0.f, 0.f, 0.f};
        #pragma unroll
        for (int kt = 0; kt < 2; ++kt) {
            v16bf bf = *(const v16bf*)&w1f[nt][kt][lane][0];
            acc = __builtin_amdgcn_wmma_f32_16x16x32_bf16(
                false, af[kt], false, bf, (short)0, acc, false, false);
        }
        const float bias = b1[nt * 16 + m];
        float* op = out + (rbase + 8 * half) * HID + nt * 16 + m;
        #pragma unroll
        for (int v = 0; v < 8; ++v) op[(long)v * HID] = acc[v] + bias;
    }
}

// ---------------------------------------------------------------------------
// Phase 2: sequential scan. Batch rows are independent -> 8 WGs, 16 rows each,
// no inter-WG sync. W2 fragments live in VGPRs for the whole scan; h lives in
// LDS (bf16, A-fragment layout, DOUBLE buffered -> one barrier per step).
// xproj[t] is read from d_out (prefetched, double buffered) and slice t is
// overwritten with h_seq[t].
// ---------------------------------------------------------------------------
__global__ __launch_bounds__(256) void rnn_scan_kernel(
    const float* __restrict__ h0, const float* __restrict__ W2,
    const float* __restrict__ b2, float* __restrict__ out)
{
    __shared__ __align__(32) __bf16 w2f[8][4][32][16]; // 32 KB, B-frag layout
    __shared__ __align__(32) __bf16 hf[2][4][32][16];  //  8 KB, A-frag, 2 bufs

    const int tid = threadIdx.x, lane = tid & 31, wave = tid >> 5;
    const int row0 = blockIdx.x * 16;

    // Stage W2 -> LDS B fragments (B(k,n) = W2[n][k])
    for (int idx = tid; idx < 8 * 4 * 32; idx += 256) {
        const int nt = idx >> 7, kt = (idx >> 5) & 3, l = idx & 31;
        const int n  = nt * 16 + (l & 15);
        const int kb = kt * 32;
        const float* src = W2 + n * HID + kb;
        #pragma unroll
        for (int e = 0; e < 16; ++e) w2f[nt][kt][l][e] = (__bf16)src[(e & 7) + 8 * (l >> 4) + ((e >> 3) << 4)];
    }
    // Stage h0 -> LDS A fragments (buffer 0)
    for (int idx = tid; idx < 4 * 32; idx += 256) {
        const int kt = idx >> 5, l = idx & 31;
        const int mm = l & 15;
        #pragma unroll
        for (int e = 0; e < 16; ++e) {
            const int k = (e & 7) + 8 * (l >> 4) + ((e >> 3) << 4);
            hf[0][kt][l][e] = (__bf16)h0[(row0 + mm) * HID + kt * 32 + k];
        }
    }
    __syncthreads();

    const int nt   = wave;            // 8 waves <-> 8 N tiles
    const int m    = lane & 15, half = lane >> 4;
    const int col  = nt * 16 + m;     // hid column this lane owns in C
    const float bias = b2[col];

    // W2 fragments resident in registers for all 4096 steps
    v16bf bf[4];
    #pragma unroll
    for (int kt = 0; kt < 4; ++kt) bf[kt] = *(const v16bf*)&w2f[nt][kt][lane][0];

    // Where this lane's h_new column lands in the A-fragment layout:
    const int ktp   = col >> 5;
    const int kk    = col & 31;
    const int dhalf = (kk >> 3) & 1;
    const int de    = (kk & 7) + ((kk >> 4) << 3);
    const int dl    = dhalf * 16 + 8 * half;       // + v per element below

    typedef __bf16 (*hbuf_t)[32][16];              // [kt][lane][elem]

    auto prefetch = [&](int t, float (&xp)[8]) {
        const float* p = out + ((size_t)t * BATCH + row0 + 8 * half) * HID + col;
        #pragma unroll
        for (int v = 0; v < 8; ++v) xp[v] = p[(size_t)v * HID];
    };

    auto step = [&](int t, hbuf_t rbuf, hbuf_t wbuf,
                    float (&xpc)[8], float (&xpn)[8]) {
        if (t + 1 < SEQ) prefetch(t + 1, xpn);   // hide xproj latency

        v16bf af[4];
        #pragma unroll
        for (int kt = 0; kt < 4; ++kt)
            af[kt] = *(const v16bf*)&rbuf[kt][lane][0];

        // Two independent accumulator chains -> shorter WMMA RAW chain
        v8f acc0 = {0.f, 0.f, 0.f, 0.f, 0.f, 0.f, 0.f, 0.f};
        v8f acc1 = {0.f, 0.f, 0.f, 0.f, 0.f, 0.f, 0.f, 0.f};
        acc0 = __builtin_amdgcn_wmma_f32_16x16x32_bf16(
            false, af[0], false, bf[0], (short)0, acc0, false, false);
        acc1 = __builtin_amdgcn_wmma_f32_16x16x32_bf16(
            false, af[1], false, bf[1], (short)0, acc1, false, false);
        acc0 = __builtin_amdgcn_wmma_f32_16x16x32_bf16(
            false, af[2], false, bf[2], (short)0, acc0, false, false);
        acc1 = __builtin_amdgcn_wmma_f32_16x16x32_bf16(
            false, af[3], false, bf[3], (short)0, acc1, false, false);

        float hn[8];
        #pragma unroll
        for (int v = 0; v < 8; ++v)
            hn[v] = TANHF(acc0[v] + acc1[v] + xpc[v] + bias);

        // Critical path first: h_new -> other LDS buffer (next step's A frags)
        #pragma unroll
        for (int v = 0; v < 8; ++v)
            wbuf[ktp][dl + v][de] = (__bf16)hn[v];

        // Fire-and-forget global stores of h_seq[t]
        float* op = out + ((size_t)t * BATCH + row0 + 8 * half) * HID + col;
        #pragma unroll
        for (int v = 0; v < 8; ++v) op[(size_t)v * HID] = hn[v];

        if (t == SEQ - 1) {
            float* lp = out + (size_t)SEQ * BATCH * HID
                            + (size_t)(row0 + 8 * half) * HID + col;
            #pragma unroll
            for (int v = 0; v < 8; ++v) lp[(size_t)v * HID] = hn[v];
        }
        __syncthreads();   // single barrier: writes visible for next step
    };

    float xpA[8], xpB[8];
    prefetch(0, xpA);
    for (int t = 0; t < SEQ; t += 2) {
        step(t,     hf[0], hf[1], xpA, xpB);
        step(t + 1, hf[1], hf[0], xpB, xpA);
    }
}

// ---------------------------------------------------------------------------
extern "C" void kernel_launch(void* const* d_in, const int* in_sizes, int n_in,
                              void* d_out, int out_size, void* d_ws, size_t ws_size,
                              hipStream_t stream) {
    const float* x  = (const float*)d_in[0];
    const float* h0 = (const float*)d_in[1];
    const float* W1 = (const float*)d_in[2];
    const float* b1 = (const float*)d_in[3];
    const float* W2 = (const float*)d_in[4];
    const float* b2 = (const float*)d_in[5];
    float* out = (float*)d_out;

    // Phase 1: input projection for all timesteps (writes xproj into h_seq region)
    xproj_kernel<<<dim3((SEQ * BATCH) / 128), dim3(256), 0, stream>>>(x, W1, b1, out);
    // Phase 2: recurrent scan, 8 independent WGs (16 batch rows each)
    rnn_scan_kernel<<<dim3(BATCH / 16), dim3(256), 0, stream>>>(h0, W2, b2, out);
}
